// S5Layer_43473658970331
// MI455X (gfx1250) — matrix operand order repaired and checked by reading, not compiled, not run on hardware
//
#include <hip/hip_runtime.h>
#include <hip/hip_bf16.h>

// ---------------------------------------------------------------------------
// S5 layer for MI455X (gfx1250, wave32, WMMA).
//   Bu  = u @ B_bar^T        -> bf16 WMMA GEMM  (M=32768, K=512, N=512 [re|im])
//                               A tile (f32 u) converted + staged in LDS
//   xs  = scan(Lambda_bar)   -> f32 chunked parallel scan (LC=128, NC=32)
//   out = 2*Re(xs @ C^T)+D*u -> bf16 WMMA GEMM  (M=32768, K=512 [re|im], N=512)
//                               A tile (bf16 xs) staged in LDS
//   state = xs[:, L-1, :]    -> complex64 tail of d_out
// ---------------------------------------------------------------------------

typedef __attribute__((ext_vector_type(16))) __bf16 v16bf;
typedef __attribute__((ext_vector_type(8)))  __bf16 v8bf;
typedef __attribute__((ext_vector_type(8)))  float  v8f;

constexpr int Bb  = 8, Ll = 4096, Hh = 512, Pp = 256;
constexpr int Mm  = Bb * Ll;        // 32768 rows
constexpr int N1  = 2 * Pp;         // 512  (Bu re | im)
constexpr int K1  = Hh;             // 512
constexpr int N2  = Hh;             // 512
constexpr int K2  = 2 * Pp;         // 512  (xs re | im)
constexpr int KT  = 512 / 32;       // 16 k-tiles (both GEMMs)
constexpr int LC  = 128;            // scan chunk length
constexpr int NC  = Ll / LC;        // 32 chunks
constexpr int LDA_S = 520;          // padded LDS row stride (bank rotation)

struct c32 { float re, im; };
__device__ __forceinline__ c32 cmul(c32 a, c32 b) {
  return { a.re * b.re - a.im * b.im, a.re * b.im + a.im * b.re };
}

// ---------------- precompute: Lambda_bar, scale = (Lb-1)/Lambda, power table -
__global__ void s5_prep(const float* __restrict__ lre, const float* __restrict__ lim,
                        const float* __restrict__ lstep,
                        c32* __restrict__ lam, c32* __restrict__ scale,
                        c32* __restrict__ lampow) {
  int p = threadIdx.x;
  if (p >= Pp) return;
  float st = __expf(lstep[p]);
  float ar = lre[p] * st, ai = lim[p] * st;
  float er = __expf(ar);
  c32 lb = { er * __cosf(ai), er * __sinf(ai) };
  lam[p] = lb;
  float dr = lre[p], di = lim[p];
  float inv = 1.0f / (dr * dr + di * di);
  c32 num = { lb.re - 1.0f, lb.im };
  scale[p] = { (num.re * dr + num.im * di) * inv,
               (num.im * dr - num.re * di) * inv };
  c32 cur = lb;                               // lampow[i] = Lambda_bar^(i+1)
  for (int i = 0; i < LC; ++i) { lampow[i * Pp + p] = cur; cur = cmul(cur, lb); }
}

// ---------------- weight packing (N-major / transposed for B-fragment loads) -
__global__ void s5_pack_w1(const float* __restrict__ Bsrc, const c32* __restrict__ scale,
                           __bf16* __restrict__ w1t) {           // [N1][K1]
  int idx = blockIdx.x * 256 + threadIdx.x;
  int h = idx & (K1 - 1);
  int n = idx >> 9;
  int p = (n < Pp) ? n : n - Pp;
  c32 s = scale[p];
  float br = Bsrc[(p * Hh + h) * 2 + 0];
  float bi = Bsrc[(p * Hh + h) * 2 + 1];
  float v = (n < Pp) ? (s.re * br - s.im * bi) : (s.re * bi + s.im * br);
  w1t[idx] = (__bf16)v;
}

__global__ void s5_pack_w2(const float* __restrict__ Csrc,
                           __bf16* __restrict__ w2t) {           // [N2][K2]
  int idx = blockIdx.x * 256 + threadIdx.x;
  int k = idx & (K2 - 1);
  int h = idx >> 9;
  float v = (k < Pp) ?  2.0f * Csrc[(h * Pp + k) * 2 + 0]
                     : -2.0f * Csrc[(h * Pp + (k - Pp)) * 2 + 1];
  w2t[idx] = (__bf16)v;
}

// ---------------- WMMA fragment loads ---------------------------------------
// 16-bit A 16x32: lane holds 16 contiguous K at 16*khalf
__device__ __forceinline__ v16bf ld_afrag(const __bf16* __restrict__ arow, int khalf) {
  v8bf lo = *(const v8bf*)(arow + 16 * khalf);
  v8bf hi = *(const v8bf*)(arow + 16 * khalf + 8);
  return __builtin_shufflevector(lo, hi, 0,1,2,3,4,5,6,7,8,9,10,11,12,13,14,15);
}
// 16-bit B 32x16: lane (n = lane&15) holds 16 contiguous K at 16*(lane>>4)
__device__ __forceinline__ v16bf ld_bfrag(const __bf16* __restrict__ bp) {
  v8bf lo = *(const v8bf*)(bp);
  v8bf hi = *(const v8bf*)(bp + 8);
  return __builtin_shufflevector(lo, hi, 0,1,2,3,4,5,6,7,8,9,10,11,12,13,14,15);
}

// ---------------- GEMM1: Bu = u(M x K1, f32) * w1t^T -> bu f32 (M x N1) -----
// Block: 256 thr = 8 waves; block owns 32 rows; wave owns a 64-col strip.
// A tile converted f32->bf16 and staged once in LDS, read back via ds_load.
__global__ void __launch_bounds__(256)
s5_gemm1(const float* __restrict__ U, const __bf16* __restrict__ Wt,
         float* __restrict__ Cout) {
  __shared__ __bf16 As[32 * LDA_S];
  const int mBase = blockIdx.x * 32;

  // stage: 32 x 512 f32 -> bf16 in LDS (8 elems per thread per iter)
  for (int it = 0; it < 8; ++it) {
    int idx = (it * 256 + threadIdx.x) * 8;       // 0..16383
    int row = idx >> 9, col = idx & 511;
    v8f g = *(const v8f*)(U + (size_t)(mBase + row) * K1 + col);
    *(v8bf*)(As + row * LDA_S + col) = __builtin_convertvector(g, v8bf);
  }
  __syncthreads();

  const int lane  = threadIdx.x & 31;
  const int ntB   = (threadIdx.x >> 5) * 4;       // wave id * 4 (16-col tiles)
  const int m0    = lane & 15, khalf = lane >> 4;
  v8f acc[2][4] = {};
  for (int kt = 0; kt < KT; ++kt) {
    v16bf a0 = ld_afrag(As + m0 * LDA_S + kt * 32, khalf);
    v16bf a1 = ld_afrag(As + (16 + m0) * LDA_S + kt * 32, khalf);
#pragma unroll
    for (int ni = 0; ni < 4; ++ni) {
      int n = (ntB + ni) * 16 + (lane & 15);
      v16bf bf = ld_bfrag(Wt + (size_t)n * K1 + kt * 32 + 16 * khalf);
      acc[0][ni] = __builtin_amdgcn_wmma_f32_16x16x32_bf16(
          false, a0, false, bf, (short)0, acc[0][ni], false, false);
      acc[1][ni] = __builtin_amdgcn_wmma_f32_16x16x32_bf16(
          false, a1, false, bf, (short)0, acc[1][ni], false, false);
    }
  }
  const int col0  = lane & 15;
  const int rhalf = (lane >> 4) * 8;
#pragma unroll
  for (int mi = 0; mi < 2; ++mi)
#pragma unroll
    for (int ni = 0; ni < 4; ++ni)
#pragma unroll
      for (int r = 0; r < 8; ++r) {
        int row = mBase + mi * 16 + rhalf + r;
        int col = (ntB + ni) * 16 + col0;
        Cout[(size_t)row * N1 + col] = acc[mi][ni][r];
      }
}

// ---------------- scan: local per-chunk, then carry combine, then fixup -----
__global__ void s5_scan_local(float* __restrict__ bu, const c32* __restrict__ lam,
                              c32* __restrict__ carry) {
  int tid = blockIdx.x * 256 + threadIdx.x;         // Bb*NC*Pp threads
  int p = tid & (Pp - 1);
  int c = (tid / Pp) % NC;
  int b = tid / (Pp * NC);
  c32 a = lam[p];
  c32 x = {0.f, 0.f};
  size_t row = (size_t)(b * Ll + c * LC);
  float* re = bu + row * N1 + p;
  float* im = bu + row * N1 + Pp + p;
  for (int i = 0; i < LC; ++i) {
    float br = *re, bi = *im;
    c32 nx = { a.re * x.re - a.im * x.im + br,
               a.re * x.im + a.im * x.re + bi };
    *re = nx.re; *im = nx.im;
    x = nx;
    re += N1; im += N1;
  }
  carry[(b * NC + c) * Pp + p] = x;
}

__global__ void s5_scan_carry(const c32* __restrict__ carry, const c32* __restrict__ lampow,
                              c32* __restrict__ prefix) {
  int tid = blockIdx.x * 256 + threadIdx.x;         // Bb*Pp threads
  int p = tid & (Pp - 1);
  int b = tid / Pp;
  c32 AL = lampow[(LC - 1) * Pp + p];               // Lambda_bar^LC
  c32 acc = {0.f, 0.f};
  for (int c = 0; c < NC; ++c) {
    prefix[(b * NC + c) * Pp + p] = acc;
    c32 cr = carry[(b * NC + c) * Pp + p];
    acc = { AL.re * acc.re - AL.im * acc.im + cr.re,
            AL.re * acc.im + AL.im * acc.re + cr.im };
  }
}

__global__ void s5_fixup(const float* __restrict__ bu, const c32* __restrict__ prefix,
                         const c32* __restrict__ lampow, __bf16* __restrict__ xc,
                         float* __restrict__ dout) {
  int tid = blockIdx.x * 256 + threadIdx.x;         // Mm*Pp threads
  int p = tid & (Pp - 1);
  int m = tid / Pp;
  int b = m / Ll;
  int l = m % Ll;
  int c = l / LC;
  int i = l % LC;
  c32 pre = prefix[(b * NC + c) * Pp + p];
  c32 pw  = lampow[i * Pp + p];                     // Lambda_bar^(i+1)
  float xr = bu[(size_t)m * N1 + p]      + pre.re * pw.re - pre.im * pw.im;
  float xi = bu[(size_t)m * N1 + Pp + p] + pre.re * pw.im + pre.im * pw.re;
  xc[(size_t)m * K2 + p]      = (__bf16)xr;
  xc[(size_t)m * K2 + Pp + p] = (__bf16)xi;
  if (l == Ll - 1) {                                // state, complex64 layout
    float* st = dout + (size_t)Mm * Hh + ((size_t)b * Pp + p) * 2;
    st[0] = xr; st[1] = xi;
  }
}

// ---------------- GEMM2: out = xc * w2t^T + D*u -> d_out f32 (M x H) --------
__global__ void __launch_bounds__(256)
s5_gemm2(const __bf16* __restrict__ A, const __bf16* __restrict__ Wt,
         const float* __restrict__ u, const float* __restrict__ Dv,
         float* __restrict__ Out) {
  __shared__ __bf16 As[32 * LDA_S];
  const int mBase = blockIdx.x * 32;

  // stage: 32 x 512 bf16 tile into LDS (8 elems per thread per iter)
  for (int it = 0; it < 8; ++it) {
    int idx = (it * 256 + threadIdx.x) * 8;       // 0..16383
    int row = idx >> 9, col = idx & 511;
    *(v8bf*)(As + row * LDA_S + col) =
        *(const v8bf*)(A + (size_t)(mBase + row) * K2 + col);
  }
  __syncthreads();

  const int lane  = threadIdx.x & 31;
  const int ntB   = (threadIdx.x >> 5) * 4;
  const int m0    = lane & 15, khalf = lane >> 4;
  v8f acc[2][4] = {};
  for (int kt = 0; kt < KT; ++kt) {
    v16bf a0 = ld_afrag(As + m0 * LDA_S + kt * 32, khalf);
    v16bf a1 = ld_afrag(As + (16 + m0) * LDA_S + kt * 32, khalf);
#pragma unroll
    for (int ni = 0; ni < 4; ++ni) {
      int n = (ntB + ni) * 16 + (lane & 15);
      v16bf bf = ld_bfrag(Wt + (size_t)n * K2 + kt * 32 + 16 * khalf);
      acc[0][ni] = __builtin_amdgcn_wmma_f32_16x16x32_bf16(
          false, a0, false, bf, (short)0, acc[0][ni], false, false);
      acc[1][ni] = __builtin_amdgcn_wmma_f32_16x16x32_bf16(
          false, a1, false, bf, (short)0, acc[1][ni], false, false);
    }
  }
  const int col0  = lane & 15;
  const int rhalf = (lane >> 4) * 8;
#pragma unroll
  for (int mi = 0; mi < 2; ++mi)
#pragma unroll
    for (int ni = 0; ni < 4; ++ni)
#pragma unroll
      for (int r = 0; r < 8; ++r) {
        int row = mBase + mi * 16 + rhalf + r;
        int col = (ntB + ni) * 16 + col0;
        Out[(size_t)row * N2 + col] =
            acc[mi][ni][r] + Dv[col] * u[(size_t)row * Hh + col];
      }
}

// ---------------------------------------------------------------------------
extern "C" void kernel_launch(void* const* d_in, const int* in_sizes, int n_in,
                              void* d_out, int out_size, void* d_ws, size_t ws_size,
                              hipStream_t stream) {
  const float* u     = (const float*)d_in[0];   // (B,L,H)
  const float* lre   = (const float*)d_in[1];   // (P,)
  const float* lim   = (const float*)d_in[2];   // (P,)
  const float* Bsrc  = (const float*)d_in[3];   // (P,H,2)
  const float* Csrc  = (const float*)d_in[4];   // (H,P,2)
  const float* Dv    = (const float*)d_in[5];   // (H,)
  const float* lstep = (const float*)d_in[6];   // (P,)
  float* out = (float*)d_out;                   // out (M*H) then state (B*P complex)

  char* w = (char*)d_ws;
  auto carve = [&](size_t bytes) {
    void* p = (void*)w;
    w += (bytes + 255) & ~(size_t)255;
    return p;
  };
  float*  bu     = (float*) carve(sizeof(float)  * (size_t)Mm * N1);   // 64 MB
  __bf16* xc     = (__bf16*)carve(sizeof(__bf16) * (size_t)Mm * K2);   // 32 MB
  __bf16* w1t    = (__bf16*)carve(sizeof(__bf16) * (size_t)N1 * K1);
  __bf16* w2t    = (__bf16*)carve(sizeof(__bf16) * (size_t)N2 * K2);
  c32*    lam    = (c32*)   carve(sizeof(c32) * Pp);
  c32*    scale  = (c32*)   carve(sizeof(c32) * Pp);
  c32*    lampow = (c32*)   carve(sizeof(c32) * (size_t)Pp * LC);
  c32*    carry  = (c32*)   carve(sizeof(c32) * (size_t)Bb * NC * Pp);
  c32*    prefix = (c32*)   carve(sizeof(c32) * (size_t)Bb * NC * Pp);

  s5_prep<<<1, 256, 0, stream>>>(lre, lim, lstep, lam, scale, lampow);
  s5_pack_w1<<<(N1 * K1) / 256, 256, 0, stream>>>(Bsrc, scale, w1t);
  s5_pack_w2<<<(N2 * K2) / 256, 256, 0, stream>>>(Csrc, w2t);

  s5_gemm1<<<Mm / 32, 256, 0, stream>>>(u, w1t, bu);

  s5_scan_local<<<(Bb * NC * Pp) / 256, 256, 0, stream>>>(bu, lam, carry);
  s5_scan_carry<<<(Bb * Pp) / 256, 256, 0, stream>>>(carry, lampow, prefix);
  s5_fixup<<<((size_t)Mm * Pp) / 256, 256, 0, stream>>>(bu, prefix, lampow, xc, out);

  s5_gemm2<<<Mm / 32, 256, 0, stream>>>(xc, w2t, u, Dv, out);
}
